// CrossAttention_8203387536045
// MI455X (gfx1250) — matrix operand order repaired and checked by reading, not compiled, not run on hardware
//
#include <hip/hip_runtime.h>
#include <hip/hip_bf16.h>

// ---------------------------------------------------------------------------
// CDNA5 (gfx1250) fused cross-attention:
//   K1: QKV projection  (fp32 -> bf16, x @ W^T + b, WMMA bf16)
//   K2: flash attention + softmax + LayerNorm fused (WMMA bf16, fp32 accum),
//       with double-buffered K/V tiles staged via GLOBAL_LOAD_ASYNC_TO_LDS
// ---------------------------------------------------------------------------

typedef __attribute__((ext_vector_type(8)))  float  v8f;
typedef __attribute__((ext_vector_type(8)))  __bf16 v8bf;
typedef __attribute__((ext_vector_type(16))) __bf16 v16bf;

static __device__ inline v16bf bf_cat(v8bf a, v8bf b) {
  return __builtin_shufflevector(a, b, 0,1,2,3,4,5,6,7,8,9,10,11,12,13,14,15);
}

// A-matrix fragment (16x32 bf16, M x K), source row-major, stride ld (elems).
// Layout per ISA: lane<16 -> M=lane, K = {0..7, 16..23}; lane>=16 -> K = {8..15, 24..31}
static __device__ inline v16bf lds_frag_a(const __bf16* base, int ld) {
  int lane = threadIdx.x & 31;
  int half = lane >> 4;
  int r    = lane & 15;
  const __bf16* p = base + r * ld + half * 8;
  v8bf lo = *(const v8bf*)(p);
  v8bf hi = *(const v8bf*)(p + 16);
  return bf_cat(lo, hi);
}

// B-matrix fragment (32x16 bf16, K x N), where B[k][n] = src[n][k],
// src row-major with stride ld. lane<16 -> N=lane, K = {0..7, 16..23}; lane>=16 -> K = {8..15, 24..31}
static __device__ inline v16bf lds_frag_b(const __bf16* base, int ld) {
  int lane = threadIdx.x & 31;
  int half = lane >> 4;
  int n    = lane & 15;
  const __bf16* p = base + n * ld + half * 8;
  v8bf lo = *(const v8bf*)(p);
  v8bf hi = *(const v8bf*)(p + 16);
  return bf_cat(lo, hi);
}

static __device__ inline v8f wmma_bf16(v16bf a, v16bf b, v8f c) {
  // (neg_a, A, neg_b, B, c_mod, C, reuse_a, reuse_b)
  return __builtin_amdgcn_wmma_f32_16x16x32_bf16(false, a, false, b,
                                                 (short)0, c, false, false);
}

// --- CDNA5 async global->LDS copy (ASYNCcnt-tracked), per-lane 16 bytes ---
static __device__ inline void async_ld_b128(unsigned lds_off, const void* gptr) {
  asm volatile("global_load_async_to_lds_b128 %0, %1, off"
               :: "v"(lds_off), "v"((unsigned long long)(size_t)gptr)
               : "memory");
}
static __device__ inline void wait_async_le(int n) {
  if (n == 0) asm volatile("s_wait_asynccnt 0x0" ::: "memory");
  else        asm volatile("s_wait_asynccnt 0x20" ::: "memory");
}

#define DMODEL 512
#define SEQ    2048

// ---------------------------------------------------------------------------
// Kernel 1: out = (X @ W^T + bias) * scale, cast to bf16.
// transposed==0: out[row*512+col] (row-major bf16)
// transposed==1: out[(batch*512+col)*2048 + s] (V^T layout [B, D, S])
// Block: 256 thr (8 waves), tile M=128, N=64, K-step 32.
// ---------------------------------------------------------------------------
__global__ __launch_bounds__(256)
void ca_proj_kernel(const float* __restrict__ X, const float* __restrict__ W,
                    const float* __restrict__ bias, __bf16* __restrict__ out,
                    float scale, int transposed) {
  __shared__ __bf16 aT[128 * 40];  // 128 rows x 32 k, stride 40
  __shared__ __bf16 bT[64 * 40];   // 64 rows  x 32 k, stride 40

  int tid  = threadIdx.x;
  int lane = tid & 31, w = tid >> 5;
  int half = lane >> 4, r = lane & 15;
  int m0 = blockIdx.y * 128;
  int n0 = blockIdx.x * 64;
  int wm = w & 3, wn = w >> 2;  // wave tile: 32x32 at (wm, wn)

  v8f acc[2][2] = {};

  int ar = tid >> 1;        // A loader: row 0..127
  int ac = (tid & 1) * 16;  // col 0 or 16
  int bn = tid >> 2;        // B loader: row 0..63
  int bc = (tid & 3) * 8;   // col 0,8,16,24

  for (int ko = 0; ko < DMODEL; ko += 32) {
    const float4* ap = (const float4*)(X + (size_t)(m0 + ar) * DMODEL + ko + ac);
#pragma unroll
    for (int u = 0; u < 4; ++u) {
      float4 f = ap[u];
      __bf16* dst = &aT[ar * 40 + ac + u * 4];
      dst[0] = (__bf16)f.x; dst[1] = (__bf16)f.y;
      dst[2] = (__bf16)f.z; dst[3] = (__bf16)f.w;
    }
    const float4* bp = (const float4*)(W + (size_t)(n0 + bn) * DMODEL + ko + bc);
#pragma unroll
    for (int u = 0; u < 2; ++u) {
      float4 f = bp[u];
      __bf16* dst = &bT[bn * 40 + bc + u * 4];
      dst[0] = (__bf16)f.x; dst[1] = (__bf16)f.y;
      dst[2] = (__bf16)f.z; dst[3] = (__bf16)f.w;
    }
    __syncthreads();
#pragma unroll
    for (int i = 0; i < 2; ++i) {
      v16bf aF = lds_frag_a(&aT[(wm * 32 + i * 16) * 40], 40);
#pragma unroll
      for (int j = 0; j < 2; ++j) {
        v16bf bF = lds_frag_b(&bT[(wn * 32 + j * 16) * 40], 40);
        acc[i][j] = wmma_bf16(aF, bF, acc[i][j]);
      }
    }
    __syncthreads();
  }

#pragma unroll
  for (int j = 0; j < 2; ++j) {
    int col = n0 + wn * 32 + j * 16 + r;
    float b = bias[col];
#pragma unroll
    for (int i = 0; i < 2; ++i) {
      int row0 = m0 + wm * 32 + i * 16 + half * 8;  // v=0 row; +v contiguous
      if (!transposed) {
#pragma unroll
        for (int v = 0; v < 8; ++v) {
          float val = (acc[i][j][v] + b) * scale;
          out[(size_t)(row0 + v) * DMODEL + col] = (__bf16)val;
        }
      } else {
        int batch = row0 >> 11;
        int s0    = row0 & 2047;
        v8bf pk;
#pragma unroll
        for (int v = 0; v < 8; ++v)
          pk[v] = (__bf16)((acc[i][j][v] + b) * scale);
        *(v8bf*)&out[((size_t)batch * DMODEL + col) * SEQ + s0] = pk;
      }
    }
  }
}

// ---------------------------------------------------------------------------
// Kernel 2: fused flash attention + LayerNorm.
// Block: 128 thr (4 waves). Query tile Mq=32, key tile Nk=32, 64 iterations.
// K/V tiles double-buffered in LDS, staged with async global->LDS loads.
// ---------------------------------------------------------------------------
__global__ __launch_bounds__(128)
void ca_attn_kernel(const __bf16* __restrict__ Qb, const __bf16* __restrict__ Kb,
                    const __bf16* __restrict__ Vt, const float* __restrict__ gamma,
                    const float* __restrict__ beta, float* __restrict__ out) {
  __shared__ __bf16 qs[32 * 520];        // Q tile  32 x 512 (stride 520)
  __shared__ __bf16 ks[2][32 * 520];     // K tile  32 x 512, double-buffered
  __shared__ __bf16 vts[2][512 * 40];    // V^T tile 512 x 32 (stride 40), dbl-buf
  __shared__ float  ss[32 * 33];         // scores  32 x 32
  __shared__ __bf16 ps[32 * 40];         // P bf16  32 x 32 (stride 40)
  __shared__ float  mrow[32], lrow[32], resc[32];
  __shared__ float  red1[32 * 4], red2[32 * 4];
  __shared__ float  arow[32], crow[32];

  int tid  = threadIdx.x;
  int lane = tid & 31, w = tid >> 5;
  int half = lane >> 4, r = lane & 15;
  int q0    = blockIdx.x * 32;
  int batch = blockIdx.y;

  const __bf16* Kroot = Kb + (size_t)batch * SEQ * DMODEL;
  const __bf16* Vroot = Vt + (size_t)batch * DMODEL * SEQ;

  // Issue async staging of K/V tile `t` (32 keys) into buffer `buf`.
  auto issue_tile = [&](int t, int buf) {
    const __bf16* Kbase = Kroot + (size_t)(t * 32) * DMODEL;
#pragma unroll
    for (int it = 0; it < 16; ++it) {
      int idx = tid + it * 128;
      int row = idx >> 6, c8 = idx & 63;
      async_ld_b128((unsigned)(size_t)&ks[buf][row * 520 + c8 * 8],
                    Kbase + row * DMODEL + c8 * 8);
    }
    const __bf16* Vbase = Vroot + t * 32;
#pragma unroll
    for (int it = 0; it < 16; ++it) {
      int idx = tid + it * 128;
      int row = idx >> 2, c8 = idx & 3;
      async_ld_b128((unsigned)(size_t)&vts[buf][row * 40 + c8 * 8],
                    Vbase + (size_t)row * SEQ + c8 * 8);
    }
  };

  // Prologue: start tile 0, load Q tile with normal loads, init stats.
  issue_tile(0, 0);
  const __bf16* Qbase = Qb + ((size_t)batch * SEQ + q0) * DMODEL;
  for (int idx = tid; idx < 32 * 64; idx += 128) {
    int row = idx >> 6, c8 = idx & 63;
    *(v8bf*)&qs[row * 520 + c8 * 8] = *(const v8bf*)&Qbase[row * DMODEL + c8 * 8];
  }
  if (tid < 32) { mrow[tid] = -1e30f; lrow[tid] = 0.f; }

  v8f acc[2][8] = {};                 // O slice: 2 row-groups x 8 col-tiles
  int rg = (w & 1) * 16, cg = (w >> 1) * 16;
  int d0 = w * 128;
  int srow = tid >> 2, sub = tid & 3; // softmax: 4 threads per row

  for (int kt = 0; kt < 64; ++kt) {
    int cur = kt & 1;
    if (kt + 1 < 64) {
      issue_tile(kt + 1, cur ^ 1);  // prefetch next tile (async)
      wait_async_le(32);            // tile kt's 32 ops have landed (in-order)
    } else {
      wait_async_le(0);
    }
    __syncthreads();                // tile kt visible to all waves (+ Q on kt==0)

    // --- S = Q K^T (each wave: one 16x16 tile, K-depth 512) ---
    v8f sacc = {};
#pragma unroll
    for (int kk = 0; kk < DMODEL; kk += 32) {
      v16bf aF = lds_frag_a(&qs[rg * 520 + kk], 520);
      v16bf bF = lds_frag_b(&ks[cur][cg * 520 + kk], 520);
      sacc = wmma_bf16(aF, bF, sacc);
    }
#pragma unroll
    for (int v = 0; v < 8; ++v)
      ss[(rg + half * 8 + v) * 33 + cg + r] = sacc[v];
    __syncthreads();

    // --- online softmax over the 32 new columns ---
    int sb = srow * 33 + sub * 8;
    float lmax = -1e30f;
#pragma unroll
    for (int j = 0; j < 8; ++j) lmax = fmaxf(lmax, ss[sb + j]);
    red1[srow * 4 + sub] = lmax;
    __syncthreads();
    float tmax = fmaxf(fmaxf(red1[srow * 4 + 0], red1[srow * 4 + 1]),
                       fmaxf(red1[srow * 4 + 2], red1[srow * 4 + 3]));
    float m_old = mrow[srow];
    float m_new = fmaxf(m_old, tmax);
    float rs    = __expf(m_old - m_new);
    float psum  = 0.f;
#pragma unroll
    for (int j = 0; j < 8; ++j) {
      float p = __expf(ss[sb + j] - m_new);
      psum += p;
      ps[srow * 40 + sub * 8 + j] = (__bf16)p;
    }
    red2[srow * 4 + sub] = psum;
    __syncthreads();
    if (sub == 0) {
      float t = red2[srow * 4 + 0] + red2[srow * 4 + 1] +
                red2[srow * 4 + 2] + red2[srow * 4 + 3];
      lrow[srow] = lrow[srow] * rs + t;
      mrow[srow] = m_new;
      resc[srow] = rs;
    }
    __syncthreads();

    // --- rescale O, then O += P @ V (K-depth 32) ---
#pragma unroll
    for (int mg = 0; mg < 2; ++mg) {
      float rsv[8];
#pragma unroll
      for (int v = 0; v < 8; ++v) rsv[v] = resc[mg * 16 + half * 8 + v];
      v16bf aP = lds_frag_a(&ps[(mg * 16) * 40], 40);
#pragma unroll
      for (int ng = 0; ng < 8; ++ng) {
#pragma unroll
        for (int v = 0; v < 8; ++v) acc[mg][ng][v] *= rsv[v];
        v16bf bV = lds_frag_b(&vts[cur][(d0 + ng * 16) * 40], 40);
        acc[mg][ng] = wmma_bf16(aP, bV, acc[mg][ng]);
      }
    }
    __syncthreads();  // all reads of buf `cur` done before it is overwritten
  }

  // --- epilogue: 1/l normalization folded into LayerNorm ---
#pragma unroll
  for (int mg = 0; mg < 2; ++mg) {
#pragma unroll
    for (int v = 0; v < 8; ++v) {
      float s1 = 0.f, s2 = 0.f;
#pragma unroll
      for (int ng = 0; ng < 8; ++ng) {
        float x = acc[mg][ng][v];
        s1 += x; s2 += x * x;
      }
#pragma unroll
      for (int off = 1; off < 16; off <<= 1) {
        s1 += __shfl_xor(s1, off);
        s2 += __shfl_xor(s2, off);
      }
      if (r == 0) {
        int row = mg * 16 + half * 8 + v;
        red1[row * 4 + w] = s1;
        red2[row * 4 + w] = s2;
      }
    }
  }
  __syncthreads();
  if (tid < 32) {
    float s1 = red1[tid * 4 + 0] + red1[tid * 4 + 1] +
               red1[tid * 4 + 2] + red1[tid * 4 + 3];
    float s2 = red2[tid * 4 + 0] + red2[tid * 4 + 1] +
               red2[tid * 4 + 2] + red2[tid * 4 + 3];
    float inv_l = 1.f / lrow[tid];
    float mu  = s1 * inv_l * (1.f / 512.f);
    float e2  = s2 * inv_l * inv_l * (1.f / 512.f);
    float var = e2 - mu * mu;
    float rstd = rsqrtf(var + 1e-5f);
    arow[tid] = rstd * inv_l;   // z_norm = O*a + c
    crow[tid] = -mu * rstd;
  }
  __syncthreads();

  float* obase = out + ((size_t)batch * SEQ + q0) * DMODEL;
#pragma unroll
  for (int mg = 0; mg < 2; ++mg) {
    float av[8], cv[8];
#pragma unroll
    for (int v = 0; v < 8; ++v) {
      int row = mg * 16 + half * 8 + v;
      av[v] = arow[row]; cv[v] = crow[row];
    }
#pragma unroll
    for (int ng = 0; ng < 8; ++ng) {
      int col = d0 + ng * 16 + r;
      float g = gamma[col], be = beta[col];
#pragma unroll
      for (int v = 0; v < 8; ++v) {
        int row = mg * 16 + half * 8 + v;
        obase[(size_t)row * DMODEL + col] = (acc[mg][ng][v] * av[v] + cv[v]) * g + be;
      }
    }
  }
}

// ---------------------------------------------------------------------------
// Host launcher
// d_in order: z_q, z_k, z_v, Wq, bq, Wk, bk, Wv, bv, ln_gamma, ln_beta
// ---------------------------------------------------------------------------
extern "C" void kernel_launch(void* const* d_in, const int* in_sizes, int n_in,
                              void* d_out, int out_size, void* d_ws, size_t ws_size,
                              hipStream_t stream) {
  const float* z_q = (const float*)d_in[0];
  const float* z_k = (const float*)d_in[1];
  const float* z_v = (const float*)d_in[2];
  const float* Wq  = (const float*)d_in[3];
  const float* bq  = (const float*)d_in[4];
  const float* Wk  = (const float*)d_in[5];
  const float* bk  = (const float*)d_in[6];
  const float* Wv  = (const float*)d_in[7];
  const float* bv  = (const float*)d_in[8];
  const float* ln_gamma = (const float*)d_in[9];
  const float* ln_beta  = (const float*)d_in[10];

  const size_t elems = (size_t)8 * SEQ * DMODEL;  // 8,388,608 bf16 each
  __bf16* qb = (__bf16*)d_ws;
  __bf16* kb = qb + elems;
  __bf16* vt = kb + elems;

  const float qscale = 0.04419417382415922f;  // 1/sqrt(512)

  dim3 gproj(DMODEL / 64, (8 * SEQ) / 128);   // (8, 128)
  ca_proj_kernel<<<gproj, 256, 0, stream>>>(z_q, Wq, bq, qb, qscale, 0);
  ca_proj_kernel<<<gproj, 256, 0, stream>>>(z_k, Wk, bk, kb, 1.0f, 0);
  ca_proj_kernel<<<gproj, 256, 0, stream>>>(z_v, Wv, bv, vt, 1.0f, 1);

  dim3 gattn(SEQ / 32, 8);                    // (64, 8)
  ca_attn_kernel<<<gattn, 128, 0, stream>>>(qb, kb, vt, ln_gamma, ln_beta,
                                            (float*)d_out);
}
